// Pool_85066122265500
// MI455X (gfx1250) — matrix operand order repaired and checked by reading, not compiled, not run on hardware
//
#include <hip/hip_runtime.h>
#include <hip/hip_bf16.h>
#include <stdint.h>

typedef __attribute__((ext_vector_type(16))) __bf16 v16bf;
typedef __attribute__((ext_vector_type(8)))  float  v8f;

#define N_NODES 50000
#define N_EDGES 500000
#define DIM     128
#define K_KEEP  25000
#define SORT1   65536
#define SORT2   32768

__device__ __forceinline__ float waveReduceSum(float v) {
#pragma unroll
  for (int m = 16; m >= 1; m >>= 1) v += __shfl_xor(v, m, 32);
  return v;
}

// ---------------------------------------------------------------------------
// K0: composite weights, stored TRANSPOSED for contiguous WMMA B-fragments:
//     cwT[u][j][k] = C_u[k][j],  C_m = M_m @ Wk^T (m=0..3: A0,A1,B0,B1),
//     C_4 = Wq^T  (so cwT[4][j][k] = Wq[j][k]).
//     Also v-vectors wv_m = M_m @ Wv^T (f32).
// ---------------------------------------------------------------------------
__global__ void build_composites(const float* __restrict__ Wb,
                                 const float* __restrict__ Wk,
                                 const float* __restrict__ Wq,
                                 const float* __restrict__ Wv,
                                 __bf16* __restrict__ cwT,
                                 float* __restrict__ wv) {
  int tid = blockIdx.x * blockDim.x + threadIdx.x;
  if (tid < 5 * DIM * DIM) {
    int m = tid / (DIM * DIM);
    int r = tid % (DIM * DIM);
    int k = r / DIM, j = r % DIM;
    float acc;
    if (m < 4) {
      const float* Mrow = Wb + (size_t)(m & 1) * 2 * DIM * DIM
                             + (size_t)((m >= 2 ? DIM : 0) + k) * DIM;
      acc = 0.f;
      for (int t = 0; t < DIM; ++t) acc += Mrow[t] * Wk[j * DIM + t];
    } else {
      acc = Wq[j * DIM + k];               // C_4[k][j] = Wq[j][k]
    }
    cwT[(size_t)m * DIM * DIM + (size_t)j * DIM + k] = (__bf16)acc;  // transposed
  } else {
    int idx = tid - 5 * DIM * DIM;
    if (idx < 4 * DIM) {
      int m = idx / DIM, k = idx % DIM;
      const float* Mrow = Wb + (size_t)(m & 1) * 2 * DIM * DIM
                             + (size_t)((m >= 2 ? DIM : 0) + k) * DIM;
      float acc = 0.f;
      for (int t = 0; t < DIM; ++t) acc += Mrow[t] * Wv[t];
      wv[idx] = acc;
    }
  }
}

// ---------------------------------------------------------------------------
// K1: node GEMMs via v_wmma_f32_16x16x32_bf16.
//     NF[u] = h @ C_u  for u = 0..4 (KA0,KA1,KB0,KB1,NQ), each [N,128] f32.
//     Block = 256 thr = 8 waves; block owns 64 rows (4 row-tiles); wave w owns
//     cols [16w,16w+16). B-fragments cached in registers, reused over 4 row-
//     tiles -> 16 WMMA per 32KB weight stage. All fragment LDS reads are
//     contiguous 16B/32B runs (ds_load_b128).
// ---------------------------------------------------------------------------
__global__ __launch_bounds__(256) void node_gemm(const float* __restrict__ h,
                                                 const __bf16* __restrict__ cwT,
                                                 const float* __restrict__ bq,
                                                 float* __restrict__ NF) {
  __shared__ __bf16 hA[64][DIM + 8];      // 64 rows of h, bf16, padded
  __shared__ __bf16 Ut[DIM][DIM + 8];     // Ut[j][k] = C[k][j]
  const int tid = threadIdx.x;
  const int r0  = blockIdx.x * 64;
#pragma unroll
  for (int it = 0; it < 32; ++it) {
    int idx = it * 256 + tid;             // 0..8191
    int row = idx >> 7, col = idx & 127;
    if (r0 + row < N_NODES)
      hA[row][col] = (__bf16)h[(size_t)(r0 + row) * DIM + col];
  }
  const int wave = tid >> 5, lane = tid & 31;
  const int mrow = lane & 15, hi = lane >> 4;
  const int ncol = wave * 16 + mrow;
  for (int u = 0; u < 5; ++u) {
    __syncthreads();                      // hA ready / Ut reads from prev u done
    const uint32_t* cw32 = (const uint32_t*)(cwT + (size_t)u * DIM * DIM);
#pragma unroll
    for (int it = 0; it < 32; ++it) {     // 8192 dwords = 128x128 bf16
      int idx = it * 256 + tid;
      int row = idx >> 6, c2 = (idx & 63) << 1;
      *(uint32_t*)&Ut[row][c2] = cw32[idx];
    }
    __syncthreads();
    // B 32x16 bf16: lane%16 = column N; k order used here (same as A): lanes 16..31 -> K+16.
    // b[kk][i] = C[kk*32 + hi*16 + i][ncol] = Ut[ncol][kk*32 + hi*16 + i]
    v16bf bf[4];
#pragma unroll
    for (int kk = 0; kk < 4; ++kk)
#pragma unroll
      for (int i = 0; i < 16; ++i)
        bf[kk][i] = Ut[ncol][kk * 32 + hi * 16 + i];
    const float bias = (u == 4) ? bq[ncol] : 0.f;
    float* out = NF + (size_t)u * N_NODES * DIM;
    for (int rt = 0; rt < 4; ++rt) {
      const int rbase = r0 + rt * 16;
      if (rbase >= N_NODES) break;        // block-uniform: EXEC stays all-ones
      v8f acc = {0.f, 0.f, 0.f, 0.f, 0.f, 0.f, 0.f, 0.f};
#pragma unroll
      for (int kk = 0; kk < 4; ++kk) {
        v16bf a;
#pragma unroll
        for (int i = 0; i < 16; ++i) {
          // A k order used here (same as B): lane<16 -> K = i ; lanes 16..31 -> K = i+16
          int ka = kk * 32 + hi * 16 + i;
          a[i] = hA[rt * 16 + mrow][ka];
        }
        acc = __builtin_amdgcn_wmma_f32_16x16x32_bf16(false, a, false, bf[kk],
                                                      (short)0, acc, false, false);
      }
#pragma unroll
      for (int r = 0; r < 8; ++r) {       // C/D layout: VGPR r -> M = r + 8*hi
        int M = r + hi * 8;
        out[(size_t)(rbase + M) * DIM + ncol] = acc[r] + bias;
      }
    }
  }
}

// ---------------------------------------------------------------------------
// K2: per-node scalars: vAB[m][n] = h[n].wv_m, bkq[n] = bk.nq[n],
//     selfy[n] = c4_0(vA0+vB0)+c4_1(vA1+vB1)+bv; zero segment accumulators.
// ---------------------------------------------------------------------------
__global__ void node_scalars(const float* __restrict__ h,
                             const float* __restrict__ NF,
                             const float* __restrict__ wv,
                             const float* __restrict__ bk,
                             const float* __restrict__ coeff,
                             const float* __restrict__ bv,
                             float* __restrict__ vAB,
                             float* __restrict__ bkq,
                             float* __restrict__ selfy,
                             float* __restrict__ accum) {
  const int wave = threadIdx.x >> 5, lane = threadIdx.x & 31;
  const int node = blockIdx.x * 8 + wave;
  if (node >= N_NODES) return;
  const float* NQ = NF + (size_t)4 * N_NODES * DIM;
  const float4 h4  = *(const float4*)(h  + (size_t)node * DIM + lane * 4);
  const float4 nq4 = *(const float4*)(NQ + (size_t)node * DIM + lane * 4);
  float d[5];
#pragma unroll
  for (int m = 0; m < 4; ++m) {
    const float4 w4 = *(const float4*)(wv + m * DIM + lane * 4);
    d[m] = h4.x * w4.x + h4.y * w4.y + h4.z * w4.z + h4.w * w4.w;
  }
  const float4 b4 = *(const float4*)(bk + lane * 4);
  d[4] = nq4.x * b4.x + nq4.y * b4.y + nq4.z * b4.z + nq4.w * b4.w;
#pragma unroll
  for (int m = 0; m < 5; ++m) d[m] = waveReduceSum(d[m]);
  if (lane == 0) {
#pragma unroll
    for (int m = 0; m < 4; ++m) vAB[(size_t)m * N_NODES + node] = d[m];
    bkq[node]   = d[4];
    selfy[node] = coeff[8] * (d[0] + d[2]) + coeff[9] * (d[1] + d[3]) + bv[0];
#pragma unroll
    for (int m = 0; m < 4; ++m) accum[(size_t)m * N_NODES + node] = 0.f;
  }
}

// ---------------------------------------------------------------------------
// K3: edge pass (both directions). One wave32 per edge; float4 per lane;
//     128-d dots + shuffle reduce; atomic segment sums (arrays L2-resident).
// ---------------------------------------------------------------------------
__global__ void edge_pass(const int* __restrict__ src, const int* __restrict__ dst,
                          const int* __restrict__ et,
                          const float* __restrict__ coeff,
                          const float* __restrict__ NF,
                          const float* __restrict__ vAB,
                          const float* __restrict__ bkq,
                          const float* __restrict__ bv,
                          float* __restrict__ accum) {
  const int wave = threadIdx.x >> 5, lane = threadIdx.x & 31;
  const int e = blockIdx.x * 8 + wave;
  if (e >= N_EDGES) return;
  const int s = src[e], d = dst[e], t = et[e];
  const float c0 = coeff[2 * t], c1 = coeff[2 * t + 1];
  const size_t off = (size_t)lane * 4;
  const float* KA0 = NF;
  const float* KA1 = NF + (size_t)1 * N_NODES * DIM;
  const float* KB0 = NF + (size_t)2 * N_NODES * DIM;
  const float* KB1 = NF + (size_t)3 * N_NODES * DIM;
  const float* NQ  = NF + (size_t)4 * N_NODES * DIM;
#define LD4(P, n) (*(const float4*)((P) + (size_t)(n) * DIM + off))
  float4 a0s = LD4(KA0, s), a1s = LD4(KA1, s);
  float4 b0d = LD4(KB0, d), b1d = LD4(KB1, d), qd = LD4(NQ, d);
  float4 a0d = LD4(KA0, d), a1d = LD4(KA1, d);
  float4 b0s = LD4(KB0, s), b1s = LD4(KB1, s), qs = LD4(NQ, s);
#undef LD4
  float pA = 0.f, pB = 0.f;
  pA += (c0 * (a0s.x + b0d.x) + c1 * (a1s.x + b1d.x)) * qd.x;
  pA += (c0 * (a0s.y + b0d.y) + c1 * (a1s.y + b1d.y)) * qd.y;
  pA += (c0 * (a0s.z + b0d.z) + c1 * (a1s.z + b1d.z)) * qd.z;
  pA += (c0 * (a0s.w + b0d.w) + c1 * (a1s.w + b1d.w)) * qd.w;
  pB += (c0 * (a0d.x + b0s.x) + c1 * (a1d.x + b1s.x)) * qs.x;
  pB += (c0 * (a0d.y + b0s.y) + c1 * (a1d.y + b1s.y)) * qs.y;
  pB += (c0 * (a0d.z + b0s.z) + c1 * (a1d.z + b1s.z)) * qs.z;
  pB += (c0 * (a0d.w + b0s.w) + c1 * (a1d.w + b1s.w)) * qs.w;
  pA = waveReduceSum(pA);
  pB = waveReduceSum(pB);
  if (lane == 0) {
    const float rsc = 0.08838834764831845f;  // 1/sqrt(128)
    float sA = expf(fminf(fmaxf((pA + bkq[d]) * rsc, -10.f), 10.f));
    float sB = expf(fminf(fmaxf((pB + bkq[s]) * rsc, -10.f), 10.f));
    const float* vA0 = vAB;
    const float* vA1 = vAB + (size_t)1 * N_NODES;
    const float* vB0 = vAB + (size_t)2 * N_NODES;
    const float* vB1 = vAB + (size_t)3 * N_NODES;
    float nvA = c0 * (vA0[s] + vB0[d]) + c1 * (vA1[s] + vB1[d]) + bv[0];
    float nvB = c0 * (vA0[d] + vB0[s]) + c1 * (vA1[d] + vB1[s]) + bv[0];
    atomicAdd(&accum[d], sA * nvA);                        // wV_in  @ dst
    atomicAdd(&accum[(size_t)N_NODES + d], sA);            // z_in   @ dst
    atomicAdd(&accum[(size_t)2 * N_NODES + s], sB * nvB);  // wV_out @ src
    atomicAdd(&accum[(size_t)3 * N_NODES + s], sB);        // z_out  @ src
  }
}

// ---------------------------------------------------------------------------
// K4: y = wVin/(zin+eps) + wVout/(zout+eps) + selfy; sort key (desc y, asc id)
// ---------------------------------------------------------------------------
__global__ void finalize_y(const float* __restrict__ accum,
                           const float* __restrict__ selfy,
                           float* __restrict__ y,
                           unsigned long long* __restrict__ keys1) {
  int n = blockIdx.x * blockDim.x + threadIdx.x;
  if (n >= SORT1) return;
  if (n < N_NODES) {
    float wi = accum[n],                       zi = accum[(size_t)N_NODES + n];
    float wo = accum[(size_t)2 * N_NODES + n], zo = accum[(size_t)3 * N_NODES + n];
    float v = wi / (zi + 1e-6f) + wo / (zo + 1e-6f) + selfy[n];
    y[n] = v;
    uint32_t b = __float_as_uint(v);
    uint32_t u = b ^ ((b >> 31) ? 0xFFFFFFFFu : 0x80000000u);  // order-monotone
    keys1[n] = ((unsigned long long)(~u) << 32) | (uint32_t)n;
  } else {
    keys1[n] = ~0ull;
  }
}

// ---------------------------------------------------------------------------
// Bitonic sort: global steps for j>1024, LDS-fused steps for j<=1024.
// ---------------------------------------------------------------------------
__global__ void bitonic_global(unsigned long long* __restrict__ keys, int j, int k) {
  int i = blockIdx.x * blockDim.x + threadIdx.x;
  int ixj = i ^ j;
  if (ixj > i) {
    unsigned long long a = keys[i], b = keys[ixj];
    bool up = ((i & k) == 0);
    if ((a > b) == up) { keys[i] = b; keys[ixj] = a; }
  }
}

__global__ __launch_bounds__(1024) void bitonic_shared(unsigned long long* __restrict__ keys,
                                                       int k) {
  __shared__ unsigned long long s[2048];
  const int tid  = threadIdx.x;
  const int base = blockIdx.x * 2048;
  s[tid]        = keys[base + tid];
  s[tid + 1024] = keys[base + tid + 1024];
  __syncthreads();
  int j0 = (k >> 1) < 1024 ? (k >> 1) : 1024;
  for (int j = j0; j >= 1; j >>= 1) {
    int i = 2 * tid - (tid & (j - 1));
    unsigned long long a = s[i], b = s[i + j];
    bool up = (((base + i) & k) == 0);
    if ((a > b) == up) { s[i] = b; s[i + j] = a; }
    __syncthreads();
  }
  keys[base + tid]        = s[tid];
  keys[base + tid + 1024] = s[tid + 1024];
}

// ---------------------------------------------------------------------------
// K6: second key set: (node_id << 32) | rank for the top-K ranks.
// ---------------------------------------------------------------------------
__global__ void build_keys2(const unsigned long long* __restrict__ keys1,
                            unsigned long long* __restrict__ keys2) {
  int i = blockIdx.x * blockDim.x + threadIdx.x;
  if (i >= SORT2) return;
  if (i < K_KEEP) {
    uint32_t node = (uint32_t)keys1[i];
    keys2[i] = ((unsigned long long)node << 32) | (uint32_t)i;
  } else {
    keys2[i] = ~0ull;
  }
}

// ---------------------------------------------------------------------------
// K8: outputs. Row i: node = i-th smallest kept id; rank = its top-k position.
//     updated_h[i] = h[rank] * sigmoid(y[rank])  (faithful to reference bug);
//     sorted_node_ids appended after the [K,128] block.
// ---------------------------------------------------------------------------
__global__ __launch_bounds__(128) void write_output(const unsigned long long* __restrict__ keys2,
                                                    const float* __restrict__ h,
                                                    const float* __restrict__ y,
                                                    float* __restrict__ out) {
  const int i = blockIdx.x;
  const unsigned long long key = keys2[i];
  const int node = (int)(key >> 32);
  const int rank = (int)(key & 0xFFFFFFFFu);
  const float sig = 1.f / (1.f + expf(-y[rank]));
  out[(size_t)i * DIM + threadIdx.x] = h[(size_t)rank * DIM + threadIdx.x] * sig;
  if (threadIdx.x == 0) out[(size_t)K_KEEP * DIM + i] = (float)node;
}

extern "C" void kernel_launch(void* const* d_in, const int* in_sizes, int n_in,
                              void* d_out, int out_size, void* d_ws, size_t ws_size,
                              hipStream_t stream) {
  (void)in_sizes; (void)n_in; (void)out_size; (void)ws_size;
  const float* h     = (const float*)d_in[0];
  const int*   src   = (const int*)d_in[1];
  const int*   dst   = (const int*)d_in[2];
  const int*   et    = (const int*)d_in[3];
  const float* Wb    = (const float*)d_in[4];
  const float* coeff = (const float*)d_in[5];
  const float* Wq    = (const float*)d_in[6];
  const float* bq    = (const float*)d_in[7];
  const float* Wk    = (const float*)d_in[8];
  const float* bk    = (const float*)d_in[9];
  const float* Wv    = (const float*)d_in[10];
  const float* bv    = (const float*)d_in[11];

  uint8_t* ws = (uint8_t*)d_ws;
  size_t off = 0;
  auto take = [&](size_t bytes) -> uint8_t* {
    uint8_t* p = ws + off;
    off = (off + bytes + 255) & ~(size_t)255;
    return p;
  };
  __bf16* cwT  = (__bf16*)take((size_t)5 * DIM * DIM * sizeof(__bf16));
  float*  wv   = (float*) take((size_t)4 * DIM * sizeof(float));
  float*  NF   = (float*) take((size_t)5 * N_NODES * DIM * sizeof(float));
  float*  vAB  = (float*) take((size_t)4 * N_NODES * sizeof(float));
  float*  bkq  = (float*) take((size_t)N_NODES * sizeof(float));
  float*  sfy  = (float*) take((size_t)N_NODES * sizeof(float));
  float*  acc  = (float*) take((size_t)4 * N_NODES * sizeof(float));
  float*  y    = (float*) take((size_t)N_NODES * sizeof(float));
  unsigned long long* keys1 = (unsigned long long*)take((size_t)SORT1 * 8);
  unsigned long long* keys2 = (unsigned long long*)take((size_t)SORT2 * 8);

  build_composites<<<(5 * DIM * DIM + 4 * DIM + 255) / 256, 256, 0, stream>>>(
      Wb, Wk, Wq, Wv, cwT, wv);
  node_gemm<<<(N_NODES / 16 + 3) / 4, 256, 0, stream>>>(h, cwT, bq, NF);
  node_scalars<<<(N_NODES + 7) / 8, 256, 0, stream>>>(h, NF, wv, bk, coeff, bv,
                                                      vAB, bkq, sfy, acc);
  edge_pass<<<(N_EDGES + 7) / 8, 256, 0, stream>>>(src, dst, et, coeff, NF, vAB,
                                                   bkq, bv, acc);
  finalize_y<<<SORT1 / 256, 256, 0, stream>>>(acc, sfy, y, keys1);
  for (int k = 2; k <= SORT1; k <<= 1) {
    for (int j = k >> 1; j > 1024; j >>= 1)
      bitonic_global<<<SORT1 / 256, 256, 0, stream>>>(keys1, j, k);
    bitonic_shared<<<SORT1 / 2048, 1024, 0, stream>>>(keys1, k);
  }
  build_keys2<<<SORT2 / 256, 256, 0, stream>>>(keys1, keys2);
  for (int k = 2; k <= SORT2; k <<= 1) {
    for (int j = k >> 1; j > 1024; j >>= 1)
      bitonic_global<<<SORT2 / 256, 256, 0, stream>>>(keys2, j, k);
    bitonic_shared<<<SORT2 / 2048, 1024, 0, stream>>>(keys2, k);
  }
  write_output<<<K_KEEP, DIM, 0, stream>>>(keys2, h, y, (float*)d_out);
}
